// GATv2_10041633538649
// MI455X (gfx1250) — matrix operand order repaired and checked by reading, not compiled, not run on hardware
//
#include <hip/hip_runtime.h>
#include <hip/hip_bf16.h>

#define HC 256
#define HEADS 4
#define NEGS 0.2f

typedef __attribute__((ext_vector_type(16))) __bf16 v16bf;
typedef __attribute__((ext_vector_type(8)))  __bf16 v8bf;
typedef __attribute__((ext_vector_type(8)))  float  v8f;

// ---------------- utility kernels ----------------

__global__ void k_fill_u32(unsigned int* __restrict__ p, unsigned int v, int n) {
  int i = blockIdx.x * blockDim.x + threadIdx.x;
  int stride = gridDim.x * blockDim.x;
  for (; i < n; i += stride) p[i] = v;
}

__global__ void k_cvt_bf16(const float* __restrict__ in, __bf16* __restrict__ out, int n) {
  int i = blockIdx.x * blockDim.x + threadIdx.x;
  int stride = gridDim.x * blockDim.x;
  for (; i < n; i += stride) out[i] = (__bf16)in[i];
}

// weights: f32 row-major [K][N] -> bf16 transposed [N][K]
__global__ void k_cvt_bf16_T(const float* __restrict__ in, __bf16* __restrict__ out,
                             int K, int N) {
  int i = blockIdx.x * blockDim.x + threadIdx.x;
  int stride = gridDim.x * blockDim.x;
  int total = K * N;
  for (; i < total; i += stride) {
    int n = i / K, k = i - n * K;
    out[i] = (__bf16)in[(size_t)k * N + n];
  }
}

// ---------------- WMMA bf16 GEMM:  C[M,N] = A[M,K] * B[K,N] + bias ----------------
// Bt is B transposed ([N][K] bf16). Block: 256 thr = 8 waves, tile 256x64.
// Wave tile 32x64: 2 A-frags x 4 N-tiles = 8 wmma per 32-wide k-step.
// B strip (64 cols x K) staged ONCE into LDS via global_load_async_to_lds_b128,
// padded stride K+8 (col step = 4 dwords mod 64 banks -> conflict-free b128 reads).
// A frag as loaded here: lane l: row=l&15; elem e -> K=16*(l>>4)+e (contiguous 16),
// same k order as B. B frag as loaded here: lane l: col=l&15;
// elem e -> K=16*(l>>4)+e (contiguous 16). C/D: col=l&15, elem r -> row=8*(l>>4)+r.
__global__ __launch_bounds__(256) void k_gemm_bf16(
    const __bf16* __restrict__ A, const __bf16* __restrict__ Bt,
    const float* __restrict__ bias, float* __restrict__ C,
    int M, int N, int K)
{
  __shared__ __align__(16) __bf16 Bl[64 * 264];   // up to K=256, stride K+8
  const int tid  = threadIdx.x;
  const int lane = tid & 31;
  const int wv   = tid >> 5;
  const int half = lane >> 4;
  const int l16  = lane & 15;
  const int rowBase = blockIdx.x * 256 + wv * 32;
  const int nBase   = blockIdx.y * 64;
  const int KP = K + 8;

  // ---- async-stage 64 x K B-strip into LDS ----
  {
    const int total = 64 * K;
    for (int base = tid * 8; base < total; base += 256 * 8) {
      int col = base / K;
      int k   = base - col * K;              // 8-elem chunk stays in one column
      unsigned loff = (unsigned)(uintptr_t)(const void*)&Bl[col * KP + k];
      unsigned long long gaddr =
          (unsigned long long)(uintptr_t)(Bt + (size_t)(nBase + col) * K + k);
      asm volatile("global_load_async_to_lds_b128 %0, %1, off"
                   :: "v"(loff), "v"(gaddr) : "memory");
    }
    asm volatile("s_wait_asynccnt 0x0" ::: "memory");
  }
  __syncthreads();

  const int mrow0 = rowBase + l16;
  const int mrow1 = rowBase + 16 + l16;
  const __bf16* ap0 = A + (size_t)((mrow0 < M) ? mrow0 : (M - 1)) * K + half * 16;
  const __bf16* ap1 = A + (size_t)((mrow1 < M) ? mrow1 : (M - 1)) * K + half * 16;

  v8f acc[2][4];
  #pragma unroll
  for (int mi = 0; mi < 2; ++mi)
    #pragma unroll
    for (int t = 0; t < 4; ++t)
      #pragma unroll
      for (int r = 0; r < 8; ++r) acc[mi][t][r] = 0.0f;

  for (int k0 = 0; k0 < K; k0 += 32) {
    v16bf af[2];
    {
      v8bf a0 = *(const v8bf*)(ap0 + k0);
      v8bf a1 = *(const v8bf*)(ap0 + k0 + 8);
      v8bf a2 = *(const v8bf*)(ap1 + k0);
      v8bf a3 = *(const v8bf*)(ap1 + k0 + 8);
      #pragma unroll
      for (int j = 0; j < 8; ++j) {
        af[0][j] = a0[j]; af[0][8 + j] = a1[j];
        af[1][j] = a2[j]; af[1][8 + j] = a3[j];
      }
    }
    #pragma unroll
    for (int t = 0; t < 4; ++t) {
      const __bf16* bp = &Bl[(t * 16 + l16) * KP + k0 + half * 16];
      v8bf b0 = *(const v8bf*)bp;
      v8bf b1 = *(const v8bf*)(bp + 8);
      v16bf bfrag;
      #pragma unroll
      for (int j = 0; j < 8; ++j) { bfrag[j] = b0[j]; bfrag[8 + j] = b1[j]; }
      acc[0][t] = __builtin_amdgcn_wmma_f32_16x16x32_bf16(
          false, af[0], false, bfrag, (short)0, acc[0][t], false, false);
      acc[1][t] = __builtin_amdgcn_wmma_f32_16x16x32_bf16(
          false, af[1], false, bfrag, (short)0, acc[1][t], false, false);
    }
  }

  #pragma unroll
  for (int mi = 0; mi < 2; ++mi)
    #pragma unroll
    for (int t = 0; t < 4; ++t) {
      const int ncol = nBase + t * 16 + l16;
      const float bval = bias[ncol];
      #pragma unroll
      for (int r = 0; r < 8; ++r) {
        int row = rowBase + mi * 16 + half * 8 + r;
        if (row < M) C[(size_t)row * N + ncol] = acc[mi][t][r] + bval;
      }
    }
}

// ---------------- edge-phase kernels ----------------

__device__ __forceinline__ void atomic_max_f32(float* addr, float v) {
  // monotonic int encoding; segmax buffer initialized to 0xFF800000 (-inf)
  if (v >= 0.0f) atomicMax((int*)addr, __float_as_int(v));
  else           atomicMin((unsigned int*)addr, __float_as_uint(v));
}

// wave per edge: logits[e,h] = sum_c leakyrelu(xl[s]+xr[d]) * att; seg-max over dst
__global__ __launch_bounds__(256) void k_edge_logits(
    const float* __restrict__ xl, const float* __restrict__ xr,
    const int* __restrict__ src, const int* __restrict__ dst,
    const float* __restrict__ att, float* __restrict__ logits,
    float* __restrict__ segmax, int E, int E2)
{
  const int lane = threadIdx.x & 31;
  const int e = blockIdx.x * 8 + (threadIdx.x >> 5);
  if (e >= E2) return;
  int s, d;
  if (e < E) { s = src[e]; d = dst[e]; } else { s = e - E; d = s; }
  const float* xls = xl + (size_t)s * HC;
  const float* xrd = xr + (size_t)d * HC;
  __builtin_prefetch(xls, 0, 3);
  __builtin_prefetch(xrd, 0, 3);
  float acc[HEADS];
  #pragma unroll
  for (int h = 0; h < HEADS; ++h) acc[h] = 0.0f;
  #pragma unroll
  for (int c = 0; c < 8; ++c) {
    int idx = c * 32 + lane;
    float v = xls[idx] + xrd[idx];
    v = v > 0.0f ? v : NEGS * v;
    acc[c >> 1] += v * att[idx];
  }
  #pragma unroll
  for (int h = 0; h < HEADS; ++h) {
    float a = acc[h];
    #pragma unroll
    for (int off = 16; off > 0; off >>= 1) a += __shfl_xor(a, off, 32);
    if (lane == 0) {
      logits[(size_t)e * HEADS + h] = a;
      atomic_max_f32(&segmax[(size_t)d * HEADS + h], a);
    }
  }
}

// thread per (e,h): p = exp(logit - max[dst]); denom[dst] += p   (p overwrites logits)
__global__ void k_edge_exp(float* __restrict__ p,
                           const int* __restrict__ dst,
                           const float* __restrict__ segmax,
                           float* __restrict__ denom, int E, int E2)
{
  int t = blockIdx.x * blockDim.x + threadIdx.x;
  if (t >= E2 * HEADS) return;
  int e = t >> 2, h = t & 3;
  int d = (e < E) ? dst[e] : (e - E);
  float v = __expf(p[t] - segmax[(size_t)d * HEADS + h]);
  p[t] = v;
  unsafeAtomicAdd(&denom[(size_t)d * HEADS + h], v);
}

// wave per edge: agg[dst] += alpha * xl[src]
__global__ __launch_bounds__(256) void k_edge_aggregate(
    const float* __restrict__ xl, const float* __restrict__ p,
    const float* __restrict__ denom,
    const int* __restrict__ src, const int* __restrict__ dst,
    float* __restrict__ agg, int E, int E2)
{
  const int lane = threadIdx.x & 31;
  const int e = blockIdx.x * 8 + (threadIdx.x >> 5);
  if (e >= E2) return;
  int s, d;
  if (e < E) { s = src[e]; d = dst[e]; } else { s = e - E; d = s; }
  float alpha[HEADS];
  #pragma unroll
  for (int h = 0; h < HEADS; ++h)
    alpha[h] = p[(size_t)e * HEADS + h] / (denom[(size_t)d * HEADS + h] + 1e-16f);
  const float* xls = xl + (size_t)s * HC;
  float* aggd = agg + (size_t)d * HC;
  __builtin_prefetch(xls, 0, 3);
  #pragma unroll
  for (int c = 0; c < 8; ++c) {
    int idx = c * 32 + lane;
    unsafeAtomicAdd(&aggd[idx], alpha[c >> 1] * xls[idx]);
  }
}

// h = relu(agg + bias) -> bf16 (input for next WMMA GEMM)
__global__ void k_finalize(const float* __restrict__ agg, const float* __restrict__ bias,
                           __bf16* __restrict__ hb, int n)
{
  int i = blockIdx.x * blockDim.x + threadIdx.x;
  int stride = gridDim.x * blockDim.x;
  for (; i < n; i += stride) {
    float v = agg[i] + bias[i & (HC - 1)];
    hb[i] = (__bf16)(v > 0.0f ? v : 0.0f);
  }
}

// out = sigmoid(hid @ Wp2 + bp2)   (50000x64 @ 64x8 - tiny, plain VALU)
__global__ void k_post(const float* __restrict__ hid, const float* __restrict__ Wp2,
                       const float* __restrict__ bp2, float* __restrict__ out, int Nn)
{
  int t = blockIdx.x * blockDim.x + threadIdx.x;
  if (t >= Nn * 8) return;
  int n = t >> 3, o = t & 7;
  float acc = bp2[o];
  const float* h = hid + (size_t)n * 64;
  #pragma unroll
  for (int j = 0; j < 64; ++j) acc += h[j] * Wp2[j * 8 + o];
  out[t] = 1.0f / (1.0f + __expf(-acc));
}

// ---------------- launch ----------------

extern "C" void kernel_launch(void* const* d_in, const int* in_sizes, int n_in,
                              void* d_out, int out_size, void* d_ws, size_t ws_size,
                              hipStream_t stream)
{
  const float* x    = (const float*)d_in[0];
  const int*   src  = (const int*)d_in[1];
  const int*   dst  = (const int*)d_in[2];
  const float* W1l  = (const float*)d_in[3];
  const float* b1l  = (const float*)d_in[4];
  const float* W1r  = (const float*)d_in[5];
  const float* b1r  = (const float*)d_in[6];
  const float* att1 = (const float*)d_in[7];
  const float* bias1= (const float*)d_in[8];
  const float* W2l  = (const float*)d_in[9];
  const float* b2l  = (const float*)d_in[10];
  const float* W2r  = (const float*)d_in[11];
  const float* b2r  = (const float*)d_in[12];
  const float* att2 = (const float*)d_in[13];
  const float* bias2= (const float*)d_in[14];
  const float* Wp1  = (const float*)d_in[15];
  const float* bp1  = (const float*)d_in[16];
  const float* Wp2  = (const float*)d_in[17];
  const float* bp2  = (const float*)d_in[18];
  float* out = (float*)d_out;

  const int IN_DIM = 128;
  const int Nn = in_sizes[0] / IN_DIM;
  const int E  = in_sizes[1];
  const int E2 = E + Nn;

  char* ws = (char*)d_ws;
  auto al = [](size_t v) { return (v + 255) & ~(size_t)255; };
  size_t o = 0;
  size_t O_XB  = o; o = al(o + (size_t)Nn * IN_DIM * 2);
  size_t O_W1L = o; o = al(o + (size_t)IN_DIM * HC * 2);
  size_t O_W1R = o; o = al(o + (size_t)IN_DIM * HC * 2);
  size_t O_W2L = o; o = al(o + (size_t)HC * HC * 2);
  size_t O_W2R = o; o = al(o + (size_t)HC * HC * 2);
  size_t O_WP1 = o; o = al(o + (size_t)HC * 64 * 2);
  size_t O_XL  = o; o = al(o + (size_t)Nn * HC * 4);
  size_t O_XR  = o; o = al(o + (size_t)Nn * HC * 4);
  size_t O_AGG = o; o = al(o + (size_t)Nn * HC * 4);
  size_t O_HB  = o; o = al(o + (size_t)Nn * HC * 2);
  size_t O_P   = o; o = al(o + (size_t)E2 * HEADS * 4);
  size_t O_MAX = o; o = al(o + (size_t)Nn * HEADS * 4);
  size_t O_DEN = o; o = al(o + (size_t)Nn * HEADS * 4);
  size_t O_HID = o; o = al(o + (size_t)Nn * 64 * 4);

  __bf16* xb   = (__bf16*)(ws + O_XB);
  __bf16* w1lb = (__bf16*)(ws + O_W1L);
  __bf16* w1rb = (__bf16*)(ws + O_W1R);
  __bf16* w2lb = (__bf16*)(ws + O_W2L);
  __bf16* w2rb = (__bf16*)(ws + O_W2R);
  __bf16* wp1b = (__bf16*)(ws + O_WP1);
  float*  xlb  = (float*)(ws + O_XL);
  float*  xrb  = (float*)(ws + O_XR);
  float*  aggb = (float*)(ws + O_AGG);
  __bf16* hb   = (__bf16*)(ws + O_HB);
  float*  pbuf = (float*)(ws + O_P);
  float*  smax = (float*)(ws + O_MAX);
  float*  den  = (float*)(ws + O_DEN);
  float*  hid  = (float*)(ws + O_HID);

  auto cdiv = [](int a, int b) { return (a + b - 1) / b; };
  const int TB = 256;

  // x -> bf16 row-major; weights -> bf16 transposed [N][K]
  k_cvt_bf16<<<cdiv(Nn * IN_DIM, TB), TB, 0, stream>>>(x, xb, Nn * IN_DIM);
  k_cvt_bf16_T<<<cdiv(IN_DIM * HC, TB), TB, 0, stream>>>(W1l, w1lb, IN_DIM, HC);
  k_cvt_bf16_T<<<cdiv(IN_DIM * HC, TB), TB, 0, stream>>>(W1r, w1rb, IN_DIM, HC);
  k_cvt_bf16_T<<<cdiv(HC * HC, TB), TB, 0, stream>>>(W2l, w2lb, HC, HC);
  k_cvt_bf16_T<<<cdiv(HC * HC, TB), TB, 0, stream>>>(W2r, w2rb, HC, HC);
  k_cvt_bf16_T<<<cdiv(HC * 64, TB), TB, 0, stream>>>(Wp1, wp1b, HC, 64);

  auto edge_phase = [&](const float* att, const float* bias) {
    k_fill_u32<<<cdiv(Nn * HEADS, TB), TB, 0, stream>>>((unsigned int*)smax, 0xFF800000u, Nn * HEADS);
    k_fill_u32<<<cdiv(Nn * HEADS, TB), TB, 0, stream>>>((unsigned int*)den, 0u, Nn * HEADS);
    k_fill_u32<<<cdiv(Nn * HC, TB), TB, 0, stream>>>((unsigned int*)aggb, 0u, Nn * HC);
    const int eb = cdiv(E2, 8);
    k_edge_logits<<<eb, 256, 0, stream>>>(xlb, xrb, src, dst, att, pbuf, smax, E, E2);
    k_edge_exp<<<cdiv(E2 * HEADS, TB), TB, 0, stream>>>(pbuf, dst, smax, den, E, E2);
    k_edge_aggregate<<<eb, 256, 0, stream>>>(xlb, pbuf, den, src, dst, aggb, E, E2);
    k_finalize<<<cdiv(Nn * HC, TB), TB, 0, stream>>>(aggb, bias, hb, Nn * HC);
  };

  // ---- conv1 ----
  dim3 g1(cdiv(Nn, 256), HC / 64);
  k_gemm_bf16<<<g1, 256, 0, stream>>>(xb, w1lb, b1l, xlb, Nn, HC, IN_DIM);
  k_gemm_bf16<<<g1, 256, 0, stream>>>(xb, w1rb, b1r, xrb, Nn, HC, IN_DIM);
  edge_phase(att1, bias1);

  // ---- conv2 (input hb, bf16) ----
  k_gemm_bf16<<<g1, 256, 0, stream>>>(hb, w2lb, b2l, xlb, Nn, HC, HC);
  k_gemm_bf16<<<g1, 256, 0, stream>>>(hb, w2rb, b2r, xrb, Nn, HC, HC);
  edge_phase(att2, bias2);

  // ---- post_mp ----
  dim3 g3(cdiv(Nn, 256), 1);
  k_gemm_bf16<<<g3, 256, 0, stream>>>(hb, wp1b, bp1, hid, Nn, 64, HC);
  k_post<<<cdiv(Nn * 8, TB), TB, 0, stream>>>(hid, Wp2, bp2, out, Nn);
}
